// GrowingNet_14310831030700
// MI455X (gfx1250) — matrix operand order repaired and checked by reading, not compiled, not run on hardware
//
#include <hip/hip_runtime.h>

typedef __attribute__((ext_vector_type(16))) _Float16 v16h;
typedef __attribute__((ext_vector_type(8)))  float    v8f;
typedef __attribute__((ext_vector_type(4)))  float    v4f;
typedef __attribute__((ext_vector_type(2)))  _Float16 h2;

#define PIX 20   // padded floats per pixel record (16 channels + 4 pad) -> conflict-free stride

static __device__ __forceinline__ v8f wmma_f16(v16h a, v16h b, v8f c) {
  // D = A(16x32 f16) * B(32x16 f16) + C(16x16 f32)
  return __builtin_amdgcn_wmma_f32_16x16x32_f16(false, a, false, b, (short)0, c, false, false);
}

// pack two f32 to f16 pair (cvt_pk output is canonical -> no canonicalize needed downstream)
static __device__ __forceinline__ unsigned pkh(float a, float b) {
  h2 p;
  p[0] = (_Float16)a;
  p[1] = (_Float16)b;
  return __builtin_bit_cast(unsigned, p);
}

// cvt first, then one packed f16 max against zero: 2 ops/dword instead of canon+max per element
static __device__ __forceinline__ unsigned pk_relu(float a, float b) {
  h2 p;
  p[0] = (_Float16)a;
  p[1] = (_Float16)b;
  h2 z = {(_Float16)0.0f, (_Float16)0.0f};
  h2 r = __builtin_elementwise_max(p, z);
  return __builtin_bit_cast(unsigned, r);
}

// lane <-> lane^16 swap in one VALU op (identity lane selects)
static __device__ __forceinline__ unsigned xor16(unsigned v) {
  return __builtin_amdgcn_permlanex16(v, v, 0x76543210u, 0xFEDCBA98u, false, false);
}

union BFrag { v16h v; unsigned u[8]; };

__global__ __launch_bounds__(256) void nca_step_kernel(
    const float* __restrict__ state, const int* __restrict__ rmask,
    const float* __restrict__ w1, const float* __restrict__ b1,
    const float* __restrict__ w2, const float* __restrict__ b2,
    float* __restrict__ out)
{
  __shared__ float sgrid[1024 * PIX];     // 80 KB, [pixel][channel] padded, whole image
  __shared__ float xalpha[32 * 32];       // 4 KB, alpha of updated state
  __shared__ v16h  wfA1[16][32];          // 16 KB, W1 A-fragments (mc*2+ks), per lane
  __shared__ v16h  wfA2[4][32];           // 4 KB,  W2 A-fragments (ks), per lane

  const int b    = blockIdx.x;            // one image per block
  const int tid  = threadIdx.x;
  const int lane = tid & 31;
  const int wave = tid >> 5;              // 8 waves
  const int n    = lane & 15;             // pixel-in-tile / N column
  const bool lo  = (lane < 16);

  const float* simg = state + (size_t)b * 16384;

  // ---- stage state image into LDS, transposing [c][y][x] -> [pixel][c] ----
  {
    const float4* g4 = (const float4*)simg;
#pragma unroll
    for (int i = 0; i < 16; ++i) {
      int flat = i * 256 + tid;           // float4 index: c = flat>>8, 4 pixels = (flat&255)*4
      float4 v = g4[flat];
      int c   = flat >> 8;
      int pix = (flat & 255) * 4;
      sgrid[(pix + 0) * PIX + c] = v.x;
      sgrid[(pix + 1) * PIX + c] = v.y;
      sgrid[(pix + 2) * PIX + c] = v.z;
      sgrid[(pix + 3) * PIX + c] = v.w;
    }
  }

  // ---- wave 0 pre-swizzles weight fragments into LDS (16-bit A 16x32, contiguous k order) ----
  // lanes 0-15: elements 0..15 -> K=0..15 ; lanes 16-31: K=16..31
  // Bias trick: perception row K=48 is constant 1.0, W1 column 48 carries b1.
  if (wave == 0) {
    const int kb = lo ? 0 : 16;
#pragma unroll
    for (int mc = 0; mc < 8; ++mc) {
      const float* wrow = w1 + (mc * 16 + n) * 48;
#pragma unroll
      for (int ks = 0; ks < 2; ++ks) {
        v16h f;
#pragma unroll
        for (int e = 0; e < 16; ++e) {
          int v = e >> 1;
          int col = ks * 32 + kb + e;
          float wv = (col < 48) ? wrow[col] : ((col == 48) ? b1[mc * 16 + n] : 0.0f);
          f[e] = (_Float16)wv;
        }
        wfA1[mc * 2 + ks][lane] = f;
      }
    }
    const float* wrow2 = w2 + n * 128;
#pragma unroll
    for (int ks = 0; ks < 4; ++ks) {
      v16h f;
#pragma unroll
      for (int e = 0; e < 16; ++e) {
        int v = e >> 1;
        int col = ks * 32 + kb + e;
        f[e] = (_Float16)wrow2[col];
      }
      wfA2[ks][lane] = f;
    }
  }

  // layer-2 bias as C fragment (C layout: lane<16 -> M=r, lane>=16 -> M=8+r)
  v8f c2;
#pragma unroll
  for (int r = 0; r < 8; ++r) c2[r] = b2[lo ? r : 8 + r];

  __syncthreads();

  auto ld4 = [&](int cq, int yy, int xx) -> v4f {   // zero-padded quad-channel LDS read
    if ((unsigned)yy < 32u && (unsigned)xx < 32u)
      return *(const v4f*)&sgrid[(yy * 32 + xx) * PIX + cq * 4];
    v4f z = {0.0f, 0.0f, 0.0f, 0.0f};
    return z;
  };

  unsigned xvp[8][4];   // updated state per tile, packed f16 pairs (channel 2d, 2d+1)

  // ============ phase 1: perception + fused 2-layer MLP per 16-pixel tile ============
#pragma unroll
  for (int t = 0; t < 8; ++t) {
    const int p = (wave * 8 + t) * 16 + n;   // pixel index in image
    const int y = p >> 5, x = p & 31;

    // B fragments of perception: lanes 0-15 K=0..15 (sobel-x), lanes 16-31 K=16..31 (sobel-y)
    // step 2: lanes 0-15 K=32..47 (raw state); lanes 16-31 K=48 -> 1.0 (bias row), rest 0
    v16h bp0, bp1;
#pragma unroll
    for (int cq = 0; cq < 4; ++cq) {
      v4f ca = ld4(cq, y - 1, x - 1);
      v4f cb = ld4(cq, y - 1, x + 1);
      v4f cd = ld4(cq, y + 1, x - 1);
      v4f ce = ld4(cq, y + 1, x + 1);
      v4f m1 = lo ? ld4(cq, y, x - 1) : ld4(cq, y - 1, x);
      v4f m2 = lo ? ld4(cq, y, x + 1) : ld4(cq, y + 1, x);
      v4f cross = lo ? (cb - ca) + (ce - cd) : (cd - ca) + (ce - cb);
      v4f sob = cross + 2.0f * (m2 - m1);
      v4f ctr;
      if (lo) ctr = *(const v4f*)&sgrid[(y * 32 + x) * PIX + cq * 4];
      else {
        v4f z = {0.0f, 0.0f, 0.0f, 0.0f};
        if (cq == 0) z[0] = 1.0f;           // bias row K=48
        ctr = z;
      }
#pragma unroll
      for (int j = 0; j < 4; ++j) {
        bp0[cq * 4 + j] = (_Float16)sob[j];
        bp1[cq * 4 + j] = (_Float16)ctr[j];
      }
    }

    // interleaved: layer-1 M-chunk pair (= layer-2 K chunk) -> repack -> layer-2 accumulate
    v8f d2 = c2;
#pragma unroll
    for (int ks = 0; ks < 4; ++ks) {
      v8f z = {};
      v8f hA = wmma_f16(wfA1[4 * ks + 0][lane], bp0, z);
      hA     = wmma_f16(wfA1[4 * ks + 1][lane], bp1, hA);
      v8f hB = wmma_f16(wfA1[4 * ks + 2][lane], bp0, z);
      hB     = wmma_f16(wfA1[4 * ks + 3][lane], bp1, hB);

      // cvt to f16 pairs + packed ReLU, then lane<->lane+16 exchange of packed dwords
      unsigned pA[4], pB[4], sA[4], sB[4];
#pragma unroll
      for (int d = 0; d < 4; ++d) {
        pA[d] = pk_relu(hA[2 * d], hA[2 * d + 1]);
        pB[d] = pk_relu(hB[2 * d], hB[2 * d + 1]);
      }
#pragma unroll
      for (int d = 0; d < 4; ++d) {
        sA[d] = xor16(pA[d]);
        sB[d] = xor16(pB[d]);
      }
      BFrag bh;
#pragma unroll
      for (int d = 0; d < 4; ++d) {
        bh.u[d]     = lo ? pA[d] : sB[d];   // K = kbase + 2d, 2d+1
        bh.u[4 + d] = lo ? sA[d] : pB[d];   // K = kbase + 8 + 2d, 2d+1
      }
      d2 = wmma_f16(wfA2[ks][lane], bh.v, d2);
    }

    // residual update with stochastic mask (state re-read: 2x b128 from LDS)
    const float mk = (float)rmask[b * 1024 + p];
    const int cbase = lo ? 0 : 8;
    v4f sA4 = *(const v4f*)&sgrid[p * PIX + cbase];
    v4f sB4 = *(const v4f*)&sgrid[p * PIX + cbase + 4];
    float xt[8];
#pragma unroll
    for (int r = 0; r < 4; ++r) xt[r]     = sA4[r] + d2[r]     * mk;
#pragma unroll
    for (int r = 0; r < 4; ++r) xt[4 + r] = sB4[r] + d2[4 + r] * mk;
    if (lo) xalpha[p] = xt[3];              // channel 3 lives at r=3 on low lanes
#pragma unroll
    for (int d = 0; d < 4; ++d) xvp[t][d] = pkh(xt[2 * d], xt[2 * d + 1]);
  }

  __syncthreads();

  // ============ phase 2: alive gating (3x3 max-pool on both alphas) + clip + store ============
#pragma unroll
  for (int t = 0; t < 8; ++t) {
    const int p = (wave * 8 + t) * 16 + n;
    const int y = p >> 5, x = p & 31;
    float pa = -1e30f, pb = -1e30f;
#pragma unroll
    for (int dy = -1; dy <= 1; ++dy)
#pragma unroll
      for (int dx2 = -1; dx2 <= 1; ++dx2) {
        int yy = y + dy, xx = x + dx2;
        if ((unsigned)yy < 32u && (unsigned)xx < 32u) {
          pa = fmaxf(pa, sgrid[(yy * 32 + xx) * PIX + 3]);  // pre-update alpha pool
          pb = fmaxf(pb, xalpha[yy * 32 + xx]);             // post-update alpha pool
        }
      }
    const bool alive = (pa > 0.1f) && (pb > 0.1f);
#pragma unroll
    for (int d = 0; d < 4; ++d) {
      h2 pr = __builtin_bit_cast(h2, xvp[t][d]);
#pragma unroll
      for (int j = 0; j < 2; ++j) {
        int r = 2 * d + j;
        int ch = lo ? r : 8 + r;
        float v = (float)pr[j];
        float o = alive ? __builtin_amdgcn_fmed3f(v, 0.0f, 1.0f) : 0.0f;  // clip as one med3
        out[(size_t)b * 16384 + ch * 1024 + p] = o;
      }
    }
  }
}

extern "C" void kernel_launch(void* const* d_in, const int* in_sizes, int n_in,
                              void* d_out, int out_size, void* d_ws, size_t ws_size,
                              hipStream_t stream) {
  const float* state = (const float*)d_in[0];
  const int*   rmask = (const int*)d_in[1];
  const float* w1    = (const float*)d_in[2];
  const float* b1    = (const float*)d_in[3];
  const float* w2    = (const float*)d_in[4];
  const float* b2    = (const float*)d_in[5];
  float* outp = (float*)d_out;
  nca_step_kernel<<<1024, 256, 0, stream>>>(state, rmask, w1, b1, w2, b2, outp);
  (void)in_sizes; (void)n_in; (void)out_size; (void)d_ws; (void)ws_size;
}
